// ExampleModel_85194971284016
// MI455X (gfx1250) — matrix operand order repaired and checked by reading, not compiled, not run on hardware
//
#include <hip/hip_runtime.h>

typedef __attribute__((ext_vector_type(16))) __bf16 v16bf;
typedef __attribute__((ext_vector_type(8)))  __bf16 v8bf;
typedef __attribute__((ext_vector_type(8)))  float  v8f;
typedef __attribute__((ext_vector_type(4)))  unsigned int v4u;
typedef __attribute__((ext_vector_type(8)))  int    v8i;
typedef __attribute__((ext_vector_type(4)))  int    v4i;

constexpr int Bb = 4, Tt = 2048, Mm = 1024, Hh = 2048, Ee = 8;
constexpr int Ss = Bb * Tt;   // 8192 tokens
constexpr int Cc = 2048;      // capacity per expert
constexpr int LDA = 40;       // LDS row stride in bf16 (80B = 64B data + 16B pad)

// ---- TDM: DMA a 128x32 bf16 tile (row stride Mm) into LDS with 16B row padding ----
// D# per CDNA5 ISA:
//  group0: count=1 | lds_addr[63:32] | global_addr[120:64] | type=2
//  group1: data_size=2B, pad_enable, pad_interval=16dw(64B), pad_amount=4dw(16B),
//          tensor_dim0=1024, tensor_dim1=2048, tile_dim0=32, tile_dim1=128,
//          tensor_dim0_stride=1024
__device__ __forceinline__ void tdm_load_tile(unsigned lds_addr, unsigned long long gaddr) {
  v4u g0;
  g0.x = 1u;                                            // count=1, user mode
  g0.y = lds_addr;                                      // LDS byte address
  g0.z = (unsigned)(gaddr & 0xFFFFFFFFull);             // global_addr[31:0]
  g0.w = (unsigned)((gaddr >> 32) & 0x1FFFFFFull)       // global_addr[56:32]
         | (2u << 30);                                  // type=2 ("image")
  v8i g1;
  g1[0] = (1 << 16)    // data_size = 2 bytes
        | (1 << 20)    // pad_enable
        | (3 << 22)    // pad_interval: 16 DWORDs (64B = one 32-elem bf16 row)
        | (3 << 25);   // pad_amount: 4 DWORDs (16B) -> row stride 80B = LDA*2
  g1[1] = (int)(1024u << 16);          // tensor_dim0 = 1024 (bits 79:48, low half)
  g1[2] = (int)(2048u << 16);          // dim0 hi = 0 | tensor_dim1 = 2048 (low half)
  g1[3] = (int)(32u << 16);            // dim1 hi = 0 | tile_dim0 = 32
  g1[4] = 128;                         // tile_dim1 = 128, tile_dim2 = 0
  g1[5] = 1024;                        // tensor_dim0_stride = 1024 (low 32)
  g1[6] = 0;                           // stride hi | tensor_dim1_stride lo
  g1[7] = 0;
  v4i gz = {0, 0, 0, 0};
#if defined(__clang_major__) && (__clang_major__ >= 23)
  v8i gz8 = {0, 0, 0, 0, 0, 0, 0, 0};
  __builtin_amdgcn_tensor_load_to_lds(g0, g1, gz, gz, gz8, 0);
#else
  __builtin_amdgcn_tensor_load_to_lds(g0, g1, gz, gz, 0);
#endif
}

// ---------------- gating: logits, softmax, top-2 ----------------
__global__ void moe_gate(const float* __restrict__ x, const float* __restrict__ wg,
                         int* __restrict__ topi, float* __restrict__ gatev) {
  int wave = threadIdx.x >> 5, lane = threadIdx.x & 31;
  int tok = blockIdx.x * 8 + wave;
  const float* xr = x + (size_t)tok * Mm;
  float a[Ee];
#pragma unroll
  for (int e = 0; e < Ee; e++) a[e] = 0.f;
  for (int m = lane; m < Mm; m += 32) {
    float xv = xr[m];
    const float* wr = wg + m * Ee;
#pragma unroll
    for (int e = 0; e < Ee; e++) a[e] = fmaf(xv, wr[e], a[e]);
  }
#pragma unroll
  for (int e = 0; e < Ee; e++)
    for (int off = 16; off >= 1; off >>= 1) a[e] += __shfl_xor(a[e], off, 32);
  if (lane == 0) {
    float mx = a[0];
    for (int e = 1; e < Ee; e++) mx = fmaxf(mx, a[e]);
    float ex[Ee], s = 0.f;
    for (int e = 0; e < Ee; e++) { ex[e] = __expf(a[e] - mx); s += ex[e]; }
    int i0 = 0;
    for (int e = 1; e < Ee; e++) if (a[e] > a[i0]) i0 = e;
    int i1 = (i0 == 0) ? 1 : 0;
    for (int e = 0; e < Ee; e++) if (e != i0 && a[e] > a[i1]) i1 = e;
    float s0 = ex[i0] / s, s1 = ex[i1] / s;
    float inv = 1.f / (s0 + s1 + 1e-9f);
    topi[tok] = i0; topi[Ss + tok] = i1;
    gatev[tok] = s0 * inv; gatev[Ss + tok] = s1 * inv;
  }
}

// ---------------- capacity assignment (deterministic, reference order) ----------------
__global__ void moe_hist(const int* __restrict__ topi, int* __restrict__ hist) {
  int j = blockIdx.y, bx = blockIdx.x, g = j * 32 + bx;
  __shared__ int h[Ee];
  if (threadIdx.x < Ee) h[threadIdx.x] = 0;
  __syncthreads();
  int e = topi[j * Ss + bx * 256 + threadIdx.x];
  atomicAdd(&h[e], 1);
  __syncthreads();
  if (threadIdx.x < Ee) hist[g * Ee + threadIdx.x] = h[threadIdx.x];
}

__global__ void moe_scan(const int* __restrict__ hist, int* __restrict__ goff) {
  int e = threadIdx.x;
  if (e >= Ee) return;
  int run = 0;
  for (int g = 0; g < 64; g++) { goff[g * Ee + e] = run; run += hist[g * Ee + e]; }
}

__global__ void moe_pos(const int* __restrict__ topi, const int* __restrict__ goff,
                        int* __restrict__ pos) {
  int j = blockIdx.y, bx = blockIdx.x, g = j * 32 + bx;
  int tid = threadIdx.x;
  int e = topi[j * Ss + bx * 256 + tid];
  __shared__ int se[256];
  se[tid] = e;
  __syncthreads();
  int rank = 0;
  for (int t = 0; t < tid; t++) rank += (se[t] == e);
  int p = goff[g * Ee + e] + rank;
  pos[j * Ss + bx * 256 + tid] = (p < Cc) ? p : -1;
}

// ---------------- scatter tokens into per-expert buffer as bf16 hi/lo ----------------
__global__ void moe_scatter(const float* __restrict__ x, const int* __restrict__ topi,
                            const int* __restrict__ pos,
                            __bf16* __restrict__ bufhi, __bf16* __restrict__ buflo) {
  int slot = blockIdx.x;            // j*Ss + tok
  int p = pos[slot];
  if (p < 0) return;
  int e = topi[slot];
  int tok = slot & (Ss - 1);
  const float* xr = x + (size_t)tok * Mm;
  size_t dst = ((size_t)e * Cc + p) * Mm;
  for (int m = threadIdx.x; m < Mm; m += blockDim.x) {
    float v = xr[m];
    __bf16 h = (__bf16)v;
    bufhi[dst + m] = h;
    buflo[dst + m] = (__bf16)(v - (float)h);
  }
}

// ---------------- f32 -> bf16 hi/lo split for w1 ----------------
__global__ void moe_cvt(const float* __restrict__ w, __bf16* __restrict__ hi,
                        __bf16* __restrict__ lo, int n) {
  int stride = gridDim.x * blockDim.x;
  for (int i = blockIdx.x * blockDim.x + threadIdx.x; i < n; i += stride) {
    float v = w[i];
    __bf16 h = (__bf16)v;
    hi[i] = h;
    lo[i] = (__bf16)(v - (float)h);
  }
}

// ---------------- row sums of w2 over M, and b2 over M ----------------
__global__ void moe_w2rs(const float* __restrict__ w2, float* __restrict__ w2rs) {
  int lane = threadIdx.x & 31, wave = threadIdx.x >> 5;
  int row = blockIdx.x * 8 + wave;          // 0..E*H-1
  const float* r = w2 + (size_t)row * Mm;
  float s = 0.f;
  for (int m = lane; m < Mm; m += 32) s += r[m];
  for (int off = 16; off >= 1; off >>= 1) s += __shfl_xor(s, off, 32);
  if (lane == 0) w2rs[row] = s;
}

__global__ void moe_b2s(const float* __restrict__ b2, float* __restrict__ b2s) {
  int lane = threadIdx.x & 31, e = threadIdx.x >> 5;
  float s = 0.f;
  for (int m = lane; m < Mm; m += 32) s += b2[e * Mm + m];
  for (int off = 16; off >= 1; off >>= 1) s += __shfl_xor(s, off, 32);
  if (lane == 0) b2s[e] = s;
}

// ---------------- expert GEMM1 (bf16x3 WMMA, TDM-fed A tiles) + relu + w2 row-sum ----------------
__global__ __launch_bounds__(256) void moe_expert_gemm(
    const __bf16* __restrict__ Ahi_g, const __bf16* __restrict__ Alo_g,
    const __bf16* __restrict__ Bhi_g, const __bf16* __restrict__ Blo_g,
    const float* __restrict__ b1, const float* __restrict__ w2rs,
    float* __restrict__ partial) {
  int hb = blockIdx.x;   // H block (16 of 128)
  int cb = blockIdx.y;   // C block (16 of 128)
  int e  = blockIdx.z;   // expert
  __shared__ __bf16 Ath[2][128 * LDA], Atl[2][128 * LDA];  // TDM dest, double buffered
  __shared__ __bf16 Bh[128 * LDA], Bl[128 * LDA];
  __shared__ float rowpart[4][128];

  int tid = threadIdx.x;
  int lane = tid & 31, wave = tid >> 5;
  int wrow  = (wave & 1) * 64;   // 64-row slab per wave
  int wcolg = wave >> 1;         // 4 column groups
  int wcol  = wcolg * 32;        // 32-col slab per wave

  v8f zero = {};
  v8f acc[4][2];
#pragma unroll
  for (int s = 0; s < 4; s++)
#pragma unroll
    for (int c = 0; c < 2; c++) acc[s][c] = zero;

  const size_t Abase = ((size_t)e * Cc + (size_t)cb * 128) * Mm;
  const size_t Bbase = (size_t)e * Mm * Hh + (size_t)hb * 128;

  // prologue: issue TDM for first A tile (hi+lo); EXEC-ignored, wave0 only reaches here
  if (tid == 0) {
    tdm_load_tile((unsigned)reinterpret_cast<unsigned long long>(&Ath[0][0]),
                  reinterpret_cast<unsigned long long>(Ahi_g + Abase));
    tdm_load_tile((unsigned)reinterpret_cast<unsigned long long>(&Atl[0][0]),
                  reinterpret_cast<unsigned long long>(Alo_g + Abase));
  }

  for (int k0 = 0; k0 < Mm; k0 += 32) {
    int pb = (k0 >> 5) & 1;
    // pipeline: issue TDM for next A tile into the other buffer
    if (tid == 0 && k0 + 32 < Mm) {
      int nb = pb ^ 1;
      tdm_load_tile((unsigned)reinterpret_cast<unsigned long long>(&Ath[nb][0]),
                    reinterpret_cast<unsigned long long>(Ahi_g + Abase + k0 + 32));
      tdm_load_tile((unsigned)reinterpret_cast<unsigned long long>(&Atl[nb][0]),
                    reinterpret_cast<unsigned long long>(Alo_g + Abase + k0 + 32));
    }
    // B tile: w1[k0+k][h0+h] stored transposed in LDS as Bt[h][k]
    {
      int k  = tid & 31;
      int hc = tid >> 5;
      const __bf16* sH = Bhi_g + Bbase + (size_t)(k0 + k) * Hh + hc * 16;
      const __bf16* sL = Blo_g + Bbase + (size_t)(k0 + k) * Hh + hc * 16;
      __builtin_prefetch(sH + (size_t)32 * Hh, 0, 1);
#pragma unroll
      for (int i = 0; i < 16; i++) {
        Bh[(hc * 16 + i) * LDA + k] = sH[i];
        Bl[(hc * 16 + i) * LDA + k] = sL[i];
      }
    }
    // wave0: current tile's 2 TDM loads are the oldest outstanding -> tensorcnt <= 2
    if (tid < 32) {
      if (k0 + 32 < Mm) __builtin_amdgcn_s_wait_tensorcnt(2);
      else              __builtin_amdgcn_s_wait_tensorcnt(0);
    }
    __syncthreads();

    int half = lane >> 4, l16 = lane & 15;
    int akb = half * 8;     // A frag: K = akb..akb+7, akb+16..akb+23
    int bkb = half * 8;     // B frag: K = bkb..bkb+7, bkb+16..bkb+23
    v16bf afh[4], afl[4], bfh[2], bfl[2];
#pragma unroll
    for (int s = 0; s < 4; s++) {
      const __bf16* rh = &Ath[pb][(wrow + 16 * s + l16) * LDA];
      const __bf16* rl = &Atl[pb][(wrow + 16 * s + l16) * LDA];
      v8bf h0 = *reinterpret_cast<const v8bf*>(rh + akb);
      v8bf h1 = *reinterpret_cast<const v8bf*>(rh + akb + 16);
      afh[s] = __builtin_shufflevector(h0, h1, 0,1,2,3,4,5,6,7,8,9,10,11,12,13,14,15);
      v8bf l0 = *reinterpret_cast<const v8bf*>(rl + akb);
      v8bf l1 = *reinterpret_cast<const v8bf*>(rl + akb + 16);
      afl[s] = __builtin_shufflevector(l0, l1, 0,1,2,3,4,5,6,7,8,9,10,11,12,13,14,15);
    }
#pragma unroll
    for (int c = 0; c < 2; c++) {
      const __bf16* rh = Bh + (wcol + 16 * c + l16) * LDA;
      const __bf16* rl = Bl + (wcol + 16 * c + l16) * LDA;
      v8bf h0 = *reinterpret_cast<const v8bf*>(rh + bkb);
      v8bf h1 = *reinterpret_cast<const v8bf*>(rh + bkb + 16);
      bfh[c] = __builtin_shufflevector(h0, h1, 0,1,2,3,4,5,6,7,8,9,10,11,12,13,14,15);
      v8bf l0 = *reinterpret_cast<const v8bf*>(rl + bkb);
      v8bf l1 = *reinterpret_cast<const v8bf*>(rl + bkb + 16);
      bfl[c] = __builtin_shufflevector(l0, l1, 0,1,2,3,4,5,6,7,8,9,10,11,12,13,14,15);
    }
#pragma unroll
    for (int s = 0; s < 4; s++)
#pragma unroll
      for (int c = 0; c < 2; c++) {
        acc[s][c] = __builtin_amdgcn_wmma_f32_16x16x32_bf16(false, afh[s], false, bfh[c],
                                                            (short)0, acc[s][c], false, false);
        acc[s][c] = __builtin_amdgcn_wmma_f32_16x16x32_bf16(false, afh[s], false, bfl[c],
                                                            (short)0, acc[s][c], false, false);
        acc[s][c] = __builtin_amdgcn_wmma_f32_16x16x32_bf16(false, afl[s], false, bfh[c],
                                                            (short)0, acc[s][c], false, false);
      }
    __syncthreads();
  }

  // epilogue: +b1, relu, dot with w2 row-sums, per-row reduce
  int l16 = lane & 15, half = lane >> 4;
  float b1v[2], w2v[2];
#pragma unroll
  for (int c = 0; c < 2; c++) {
    int hidx = hb * 128 + wcol + 16 * c + l16;
    b1v[c] = b1[e * Hh + hidx];
    w2v[c] = w2rs[e * Hh + hidx];
  }
#pragma unroll
  for (int s = 0; s < 4; s++) {
#pragma unroll
    for (int r = 0; r < 8; r++) {
      float t = 0.f;
#pragma unroll
      for (int c = 0; c < 2; c++) {
        float v = acc[s][c][r] + b1v[c];
        v = fmaxf(v, 0.f);
        t += v * w2v[c];
      }
      t += __shfl_xor(t, 1, 32);
      t += __shfl_xor(t, 2, 32);
      t += __shfl_xor(t, 4, 32);
      t += __shfl_xor(t, 8, 32);
      if (l16 == 0) rowpart[wcolg][wrow + 16 * s + 8 * half + r] = t;
    }
  }
  __syncthreads();
  if (tid < 128) {
    float s = rowpart[0][tid] + rowpart[1][tid] + rowpart[2][tid] + rowpart[3][tid];
    partial[((size_t)e * Cc + cb * 128 + tid) * 16 + hb] = s;
  }
}

// ---------------- fixed-order reduction of H-block partials (deterministic) ----------------
__global__ void moe_rowsum(const float* __restrict__ partial, float* __restrict__ rowacc) {
  int i = blockIdx.x * blockDim.x + threadIdx.x;   // 0..E*C-1
  const float* p = partial + (size_t)i * 16;
  float s = 0.f;
#pragma unroll
  for (int h = 0; h < 16; h++) s += p[h];
  rowacc[i] = s;
}

// ---------------- combine with gates ----------------
__global__ void moe_combine(const int* __restrict__ topi, const float* __restrict__ gatev,
                            const int* __restrict__ pos, const float* __restrict__ rowacc,
                            const float* __restrict__ b2s, float* __restrict__ outsum) {
  int t = blockIdx.x * blockDim.x + threadIdx.x;
  float s = 0.f;
#pragma unroll
  for (int j = 0; j < 2; j++) {
    int p = pos[j * Ss + t];
    if (p >= 0) {
      int e = topi[j * Ss + t];
      s += gatev[j * Ss + t] * (rowacc[e * Cc + p] + b2s[e]);
    }
  }
  outsum[t] = s;
}

// ---------------- per-batch log-softmax over T ----------------
__global__ void moe_lsm(const float* __restrict__ outsum, float* __restrict__ out) {
  int b = blockIdx.x, tid = threadIdx.x;
  const float* v = outsum + b * Tt;
  __shared__ float red[256];
  float mx = -1e30f;
  for (int t = tid; t < Tt; t += 256) mx = fmaxf(mx, v[t]);
  red[tid] = mx; __syncthreads();
  for (int s = 128; s > 0; s >>= 1) { if (tid < s) red[tid] = fmaxf(red[tid], red[tid + s]); __syncthreads(); }
  mx = red[0]; __syncthreads();
  float sum = 0.f;
  for (int t = tid; t < Tt; t += 256) sum += __expf(v[t] - mx);
  red[tid] = sum; __syncthreads();
  for (int s = 128; s > 0; s >>= 1) { if (tid < s) red[tid] += red[tid + s]; __syncthreads(); }
  float lse = mx + __logf(red[0]);
  for (int t = tid; t < Tt; t += 256) out[b * Tt + t] = v[t] - lse;
}

extern "C" void kernel_launch(void* const* d_in, const int* in_sizes, int n_in,
                              void* d_out, int out_size, void* d_ws, size_t ws_size,
                              hipStream_t stream) {
  const float* x  = (const float*)d_in[0];
  const float* wg = (const float*)d_in[1];
  const float* w1 = (const float*)d_in[2];
  const float* b1 = (const float*)d_in[3];
  const float* w2 = (const float*)d_in[4];
  const float* b2 = (const float*)d_in[5];
  float* out = (float*)d_out;

  char* p = (char*)d_ws;
  auto carve = [&](size_t bytes) -> void* {
    void* r = (void*)p;
    p += (bytes + 255) & ~(size_t)255;
    return r;
  };
  int*    topi   = (int*)   carve((size_t)2 * Ss * 4);
  float*  gatev  = (float*) carve((size_t)2 * Ss * 4);
  int*    hist   = (int*)   carve(64 * Ee * 4);
  int*    goff   = (int*)   carve(64 * Ee * 4);
  int*    pos    = (int*)   carve((size_t)2 * Ss * 4);
  float*  w2rs   = (float*) carve((size_t)Ee * Hh * 4);
  float*  b2s    = (float*) carve(Ee * 4);
  float*  outsum = (float*) carve((size_t)Ss * 4);
  float*  rowacc = (float*) carve((size_t)Ee * Cc * 4);
  float*  partial= (float*) carve((size_t)Ee * Cc * 16 * 4);
  __bf16* bufhi  = (__bf16*)carve((size_t)Ee * Cc * Mm * 2);
  __bf16* buflo  = (__bf16*)carve((size_t)Ee * Cc * Mm * 2);
  __bf16* w1hi   = (__bf16*)carve((size_t)Ee * Mm * Hh * 2);
  __bf16* w1lo   = (__bf16*)carve((size_t)Ee * Mm * Hh * 2);

  moe_cvt<<<4096, 256, 0, stream>>>(w1, w1hi, w1lo, Ee * Mm * Hh);
  moe_w2rs<<<Ee * Hh / 8, 256, 0, stream>>>(w2, w2rs);
  moe_b2s<<<1, 256, 0, stream>>>(b2, b2s);
  moe_gate<<<Ss / 8, 256, 0, stream>>>(x, wg, topi, gatev);
  moe_hist<<<dim3(32, 2), 256, 0, stream>>>(topi, hist);
  moe_scan<<<1, 32, 0, stream>>>(hist, goff);
  moe_pos<<<dim3(32, 2), 256, 0, stream>>>(topi, goff, pos);
  moe_scatter<<<2 * Ss, 128, 0, stream>>>(x, topi, pos, bufhi, buflo);
  moe_expert_gemm<<<dim3(16, 16, 8), 256, 0, stream>>>(bufhi, buflo, w1hi, w1lo, b1, w2rs, partial);
  moe_rowsum<<<Ee * Cc / 256, 256, 0, stream>>>(partial, rowacc);
  moe_combine<<<Ss / 256, 256, 0, stream>>>(topi, gatev, pos, rowacc, b2s, outsum);
  moe_lsm<<<Bb, 256, 0, stream>>>(outsum, out);
}
